// CrossAttention_30313879175423
// MI455X (gfx1250) — compile-verified, matrix operand order checked by reading, not run on hardware
//
#include <hip/hip_runtime.h>
#include <hip/hip_bf16.h>
#include <stdint.h>

// ---------------------------------------------------------------------------
// CrossAttention for MI455X (gfx1250): bf16 WMMA projections + fused flash
// attention with double-buffered async global->LDS K/V staging (ASYNCcnt).
// ---------------------------------------------------------------------------

typedef __attribute__((ext_vector_type(16))) __bf16 v16bf;
typedef __attribute__((ext_vector_type(8)))  float  v8f;

#define HEADS 6
#define HD    32
#define DIMQ  192
#define NQ    256
#define NKV   576
#define BATCH 256
#define NCHUNK 9          // 576 / 64

static __device__ __forceinline__ v8f zero8() {
  v8f z;
#pragma unroll
  for (int j = 0; j < 8; ++j) z[j] = 0.0f;
  return z;
}

static __device__ __forceinline__ v8f wmma_bf16(v16bf a, v16bf b, v8f c) {
  // (neg_a, A, neg_b, B, c_mod, C, reuse_a, reuse_b)
  return __builtin_amdgcn_wmma_f32_16x16x32_bf16(false, a, false, b,
                                                 (short)0, c, false, false);
}

// A matrix 16x32 bf16, per ISA layout: lane m=lane&15, half=lane>>4,
// pair p holds K = (p/4)*16 + half*8 + (p%4)*2 + {0,1}
static __device__ __forceinline__ v16bf load_A(const __bf16* p, int lane, int ld) {
  int m = lane & 15, half = lane >> 4;
  v16bf a;
#pragma unroll
  for (int q = 0; q < 8; ++q) {
    int k = ((q >> 2) << 4) + (half << 3) + ((q & 3) << 1);
    a[2 * q]     = p[m * ld + k];
    a[2 * q + 1] = p[m * ld + k + 1];
  }
  return a;
}

// B matrix 32x16 bf16, N rows in memory:  B[k][n] = p[n*ld + k]
static __device__ __forceinline__ v16bf load_B_T(const __bf16* p, int lane, int ld) {
  int n = lane & 15, half = lane >> 4;
  v16bf b;
#pragma unroll
  for (int q = 0; q < 8; ++q) {
    int k = ((q >> 2) << 4) + (half << 3) + ((q & 3) << 1);
    b[2 * q]     = p[n * ld + k];
    b[2 * q + 1] = p[n * ld + k + 1];
  }
  return b;
}

// B matrix 32x16 bf16, K rows in memory:  B[k][n] = p[k*ld + n]
static __device__ __forceinline__ v16bf load_B_N(const __bf16* p, int lane, int ld) {
  int n = lane & 15, half = lane >> 4;
  v16bf b;
#pragma unroll
  for (int q = 0; q < 8; ++q) {
    int k = ((q >> 2) << 4) + (half << 3) + ((q & 3) << 1);
    b[2 * q]     = p[k * ld + n];
    b[2 * q + 1] = p[(k + 1) * ld + n];
  }
  return b;
}

// row-wise reductions within a 16-lane half (C/D layout: N = lane&15)
static __device__ __forceinline__ float rowmax16(float x) {
#pragma unroll
  for (int m = 8; m >= 1; m >>= 1) x = fmaxf(x, __shfl_xor(x, m, 16));
  return x;
}
static __device__ __forceinline__ float rowsum16(float x) {
#pragma unroll
  for (int m = 8; m >= 1; m >>= 1) x += __shfl_xor(x, m, 16);
  return x;
}

// async global -> LDS copy, 16 bytes per lane (GLOBAL_LOAD_ASYNC_TO_LDS_B128,
// tracked by ASYNCcnt). lds_addr = low 32 bits of generic shared pointer.
static __device__ __forceinline__ void async_copy16(uint32_t lds_addr,
                                                    const void* gaddr) {
  asm volatile("global_load_async_to_lds_b128 %0, %1, off"
               :: "v"(lds_addr), "v"(gaddr)
               : "memory");
}

// ---------------------------------------------------------------------------
__global__ void ca_f32_to_bf16(const float* __restrict__ src,
                               __bf16* __restrict__ dst, int n) {
  int i = blockIdx.x * blockDim.x + threadIdx.x;
  if (i < n) dst[i] = (__bf16)src[i];
}

// bias_out[h][q][k] = table[rpi[q][k]][h]   (batch-invariant, 3.5 MB -> L2)
__global__ void ca_bias_gather(const long long* __restrict__ rpi,
                               const float* __restrict__ table,
                               float* __restrict__ biasOut) {
  int i = blockIdx.x * blockDim.x + threadIdx.x;
  if (i >= NQ * NKV) return;
  long long idx = rpi[i];
  int qrow = i / NKV, kcol = i - qrow * NKV;
#pragma unroll
  for (int hh = 0; hh < HEADS; ++hh)
    biasOut[((size_t)hh * NQ + qrow) * NKV + kcol] = table[idx * HEADS + hh];
}

// ---------------------------------------------------------------------------
// WMMA projection: out = X @ W^T, X fp32 [b][rows][192], W bf16 [cols][192].
// Writes bf16 in [b][h][row][32] layout. 4 waves/block, one 16x16 tile each.
__global__ __launch_bounds__(128)
void ca_proj(const float* __restrict__ X, const __bf16* __restrict__ W,
             __bf16* __restrict__ out0, __bf16* __restrict__ out1,
             int rowsPerB, int kvMode) {
  __shared__ __bf16 Xs[16 * DIMQ];

  int b  = blockIdx.z;
  int rt = blockIdx.y;
  int tid = threadIdx.x;
  int wave = tid >> 5, lane = tid & 31;

  const float* Xb = X + ((size_t)b * rowsPerB + rt * 16) * DIMQ;
  for (int i = tid; i < 16 * DIMQ; i += 128) Xs[i] = (__bf16)Xb[i];
  __syncthreads();

  int colBase = (blockIdx.x * 4 + wave) * 16;
  v8f c = zero8();
#pragma unroll
  for (int kk = 0; kk < DIMQ; kk += 32) {
    v16bf a = load_A(Xs + kk, lane, DIMQ);
    v16bf w = load_B_T(W + (size_t)colBase * DIMQ + kk, lane, DIMQ);
    c = wmma_bf16(a, w, c);
  }

  int n = lane & 15, half = lane >> 4;
#pragma unroll
  for (int r = 0; r < 8; ++r) {
    int row = rt * 16 + r + 8 * half;
    int col = colBase + n;
    float v = c[r];
    if (!kvMode) {
      int hh = col >> 5, d = col & 31;
      out0[(((size_t)b * HEADS + hh) * NQ + row) * HD + d] = (__bf16)v;
    } else if (col < DIMQ) {
      int hh = col >> 5, d = col & 31;
      out0[(((size_t)b * HEADS + hh) * NKV + row) * HD + d] = (__bf16)v;
    } else {
      int c2 = col - DIMQ;
      int hh = c2 >> 5, d = c2 & 31;
      out1[(((size_t)b * HEADS + hh) * NKV + row) * HD + d] = (__bf16)v;
    }
  }
}

// ---------------------------------------------------------------------------
// Fused flash attention. One block per (b,h). 8 waves x 32 queries.
// K/V streamed through double-buffered LDS via async global->LDS copies.
__global__ __launch_bounds__(256)
void ca_attn(const __bf16* __restrict__ qbf, const __bf16* __restrict__ kbf,
             const __bf16* __restrict__ vbf, const float* __restrict__ bias,
             float* __restrict__ out) {
  __shared__ __bf16 Ks[2][64 * HD];     // 2 x 4 KB (double buffer)
  __shared__ __bf16 Vs[2][64 * HD];     // 2 x 4 KB
  __shared__ __bf16 Ps[8][16 * 32];     // 8 KB: per-wave P transpose bounce

  int bh = blockIdx.x;
  int b = bh / HEADS, hh = bh - b * HEADS;
  int tid = threadIdx.x;
  int wave = tid >> 5, lane = tid & 31;
  int n = lane & 15, half = lane >> 4;

  const __bf16* Qg = qbf + (size_t)bh * NQ * HD;
  const __bf16* Kg = kbf + (size_t)bh * NKV * HD;
  const __bf16* Vg = vbf + (size_t)bh * NKV * HD;
  const float*  Bb = bias + (size_t)hh * NQ * NKV;

  const float scale = 0.17677669529663687f;  // 1/sqrt(32)

  // per-thread LDS byte addresses for async copies (low 32 bits of generic ptr)
  uint32_t ldsK[2], ldsV[2];
#pragma unroll
  for (int j = 0; j < 2; ++j) {
    ldsK[j] = (uint32_t)(uintptr_t)(&Ks[j][0]) + tid * 16;
    ldsV[j] = (uint32_t)(uintptr_t)(&Vs[j][0]) + tid * 16;
  }

  v16bf qa[2];
  v8f o0[2], o1[2];
  float mrow[2][8], srow[2][8];
#pragma unroll
  for (int i = 0; i < 2; ++i) {
    qa[i] = load_A(Qg + (size_t)(wave * 2 + i) * 16 * HD, lane, HD);
    o0[i] = zero8();
    o1[i] = zero8();
#pragma unroll
    for (int r = 0; r < 8; ++r) { mrow[i][r] = -1e30f; srow[i][r] = 0.0f; }
  }

  // prologue: async-stage chunk 0 into buffer 0
  async_copy16(ldsK[0], (const char*)Kg + tid * 16);
  async_copy16(ldsV[0], (const char*)Vg + tid * 16);

  for (int c = 0; c < NCHUNK; ++c) {
    int cur = c & 1;
    int c0 = c * 64;
    if (c + 1 < NCHUNK) {
      // prefetch next chunk into the other buffer while we compute
      const char* gk = (const char*)(Kg + (size_t)(c0 + 64) * HD) + tid * 16;
      const char* gv = (const char*)(Vg + (size_t)(c0 + 64) * HD) + tid * 16;
      async_copy16(ldsK[cur ^ 1], gk);
      async_copy16(ldsV[cur ^ 1], gv);
      // async loads complete in order: <=2 outstanding => chunk c is in LDS
      asm volatile("s_wait_asynccnt 2" ::: "memory");
    } else {
      asm volatile("s_wait_asynccnt 0" ::: "memory");
    }
    __syncthreads();

    const __bf16* Kc = &Ks[cur][0];
    const __bf16* Vc = &Vs[cur][0];

#pragma unroll
    for (int kp = 0; kp < 2; ++kp) {     // two 32-key sub-chunks
      int kb = kp * 32;
      // B tiles hoisted: shared by both M-tiles of this wave
      v16bf bk0 = load_B_T(Kc + (size_t)kb * HD, lane, HD);
      v16bf bk1 = load_B_T(Kc + (size_t)(kb + 16) * HD, lane, HD);
      v16bf bv0 = load_B_N(Vc + (size_t)kb * HD, lane, HD);
      v16bf bv1 = load_B_N(Vc + (size_t)kb * HD + 16, lane, HD);

#pragma unroll
      for (int i = 0; i < 2; ++i) {
        int q0 = (wave * 2 + i) * 16;
        // S = scale * Q K^T + bias  (two 16-key tiles)
        v8f s0 = wmma_bf16(qa[i], bk0, zero8());
        v8f s1 = wmma_bf16(qa[i], bk1, zero8());
#pragma unroll
        for (int r = 0; r < 8; ++r) {
          int row = q0 + r + 8 * half;
          int key = c0 + kb + n;
          s0[r] = s0[r] * scale + Bb[(size_t)row * NKV + key];
          s1[r] = s1[r] * scale + Bb[(size_t)row * NKV + key + 16];
        }
        // online softmax update (per-row stats, cross-lane in 16-lane halves)
#pragma unroll
        for (int r = 0; r < 8; ++r) {
          float tmax = rowmax16(fmaxf(s0[r], s1[r]));
          float mnew = fmaxf(mrow[i][r], tmax);
          float f = __expf(mrow[i][r] - mnew);
          mrow[i][r] = mnew;
          float p0 = __expf(s0[r] - mnew);
          float p1 = __expf(s1[r] - mnew);
          srow[i][r] = srow[i][r] * f + rowsum16(p0 + p1);
          o0[i][r] *= f;
          o1[i][r] *= f;
          s0[r] = p0;
          s1[r] = p1;
        }
        // transpose P (C-layout f32) -> A-layout bf16 via per-wave LDS bounce
        __bf16* Pw = &Ps[wave][0];
#pragma unroll
        for (int r = 0; r < 8; ++r) {
          int row = r + 8 * half;
          Pw[row * 32 + n]      = (__bf16)s0[r];
          Pw[row * 32 + 16 + n] = (__bf16)s1[r];
        }
        asm volatile("s_wait_dscnt 0" ::: "memory");  // wave-local LDS fence
        v16bf pa = load_A(Pw, lane, 32);
        // O += P @ V  (contract 32 keys; two 16-dim output tiles)
        o0[i] = wmma_bf16(pa, bv0, o0[i]);
        o1[i] = wmma_bf16(pa, bv1, o1[i]);
      }
    }
    __syncthreads();   // protect LDS buffers before next-iteration async issue
  }

  // finalize: O /= rowsum, store fp32 in (b, nq, h*32+d)
#pragma unroll
  for (int i = 0; i < 2; ++i) {
    int q0 = (wave * 2 + i) * 16;
#pragma unroll
    for (int r = 0; r < 8; ++r) {
      float inv = 1.0f / srow[i][r];
      int row = q0 + r + 8 * half;
      size_t base = ((size_t)b * NQ + row) * DIMQ + hh * HD;
      out[base + n]      = o0[i][r] * inv;
      out[base + 16 + n] = o1[i][r] * inv;
    }
  }
}

// ---------------------------------------------------------------------------
extern "C" void kernel_launch(void* const* d_in, const int* in_sizes, int n_in,
                              void* d_out, int out_size, void* d_ws, size_t ws_size,
                              hipStream_t stream) {
  (void)in_sizes; (void)n_in; (void)out_size; (void)ws_size;

  const float*     x_q   = (const float*)d_in[0];
  const float*     x_kv  = (const float*)d_in[1];
  const long long* rpi   = (const long long*)d_in[2];
  const float*     Wq    = (const float*)d_in[3];
  const float*     Wkv   = (const float*)d_in[4];
  const float*     table = (const float*)d_in[5];
  float*           out   = (float*)d_out;

  // workspace layout (bytes)
  char* ws = (char*)d_ws;
  __bf16* qbf   = (__bf16*)(ws + 0);           // 256*6*256*32*2  = 25,165,824
  __bf16* kbf   = (__bf16*)(ws + 25165824);    // 256*6*576*32*2  = 56,623,104
  __bf16* vbf   = (__bf16*)(ws + 81788928);    // 56,623,104
  __bf16* wqb   = (__bf16*)(ws + 138412032);   // 192*192*2       = 73,728
  __bf16* wkvb  = (__bf16*)(ws + 138485760);   // 384*192*2       = 147,456
  float*  biasw = (float*)(ws + 138633216);    // 6*256*576*4     = 3,538,944

  ca_f32_to_bf16<<<(192 * 192 + 255) / 256, 256, 0, stream>>>(Wq, wqb, 192 * 192);
  ca_f32_to_bf16<<<(384 * 192 + 255) / 256, 256, 0, stream>>>(Wkv, wkvb, 384 * 192);
  ca_bias_gather<<<(NQ * NKV + 255) / 256, 256, 0, stream>>>(rpi, table, biasw);

  // Q projection: 256 rows/b -> 16 row tiles; 192 cols -> 12 tiles = 3 blocks x 4 waves
  ca_proj<<<dim3(3, 16, BATCH), 128, 0, stream>>>(x_q, wqb, qbf, nullptr, NQ, 0);
  // KV projection: 576 rows/b -> 36 row tiles; 384 cols -> 24 tiles = 6 blocks x 4 waves
  ca_proj<<<dim3(6, 36, BATCH), 128, 0, stream>>>(x_kv, wkvb, kbf, vbf, NKV, 1);

  // fused flash attention: one block per (b, h)
  ca_attn<<<dim3(BATCH * HEADS), 256, 0, stream>>>(qbf, kbf, vbf, biasw, out);
}
